// TransformerEncoderLayer_70248485093525
// MI455X (gfx1250) — matrix operand order repaired and checked by reading, not compiled, not run on hardware
//
#include <hip/hip_runtime.h>
#include <hip/hip_bf16.h>

// ---------------------------------------------------------------------------
// Problem constants (match the JAX reference)
// ---------------------------------------------------------------------------
#define BATCH   256
#define DM      512
#define HEADS   8
#define HD      64
#define LH      150          // history length
#define LD      200          // doc length
#define LHP     160          // history padded (multiple of 32)
#define LDP     224          // doc padded (multiple of 32)
#define UTT     50
#define PAD_IDX 1

#define USE_ASYNC_LDS 1      // gfx1250 GLOBAL_LOAD_ASYNC_TO_LDS staging

typedef __attribute__((ext_vector_type(16))) _Float16 v16h;
typedef __attribute__((ext_vector_type(8)))  float    v8f;
typedef __attribute__((address_space(3)))    _Float16 lds_f16;

// ---------------------------------------------------------------------------
// CDNA5 async global->LDS DMA (ASYNCcnt pipe), 16 bytes per lane.
// VDST = per-lane LDS byte offset, VADDR = per-lane 64-bit global address.
// ---------------------------------------------------------------------------
__device__ __forceinline__ void async_g2l_b128(const _Float16* g, _Float16* l) {
#if USE_ASYNC_LDS
  const unsigned int loff = (unsigned int)(uintptr_t)(lds_f16*)l;
  const unsigned long long ga = (unsigned long long)(uintptr_t)g;
  asm volatile("global_load_async_to_lds_b128 %0, %1, off"
               :: "v"(loff), "v"(ga) : "memory");
#else
  const short4 v = *(const short4*)g;
  *(short4*)l = v;
#endif
}

__device__ __forceinline__ void wait_async_all() {
#if USE_ASYNC_LDS
  asm volatile("s_wait_asynccnt 0x0" ::: "memory");
#endif
}

// ---------------------------------------------------------------------------
// WMMA fragment loaders — exact CDNA5 wave32 VGPR layouts (ISA 7.12.2)
// ---------------------------------------------------------------------------
// A (16x32 f16, row-major, lda elems): lane holds row m=lane&15.
// lanes 0-15: K = {0..7, 16..23}; lanes 16-31: K = {8..15, 24..31}
// -> two contiguous 8-half (16 B) chunks per lane.
__device__ __forceinline__ v16h load_a_frag(const _Float16* __restrict__ A,
                                            int lda, int lane) {
  const int m   = lane & 15;
  const int klo = (lane & 16) ? 8 : 0;
  const _Float16* row = A + (size_t)m * lda;
  v16h a;
#pragma unroll
  for (int i = 0; i < 8; ++i) a[i] = row[klo + i];           // K = klo..klo+7
#pragma unroll
  for (int i = 0; i < 8; ++i) a[8 + i] = row[16 + klo + i];  // K = 16+klo..
  return a;
}

// B (32x16) where the source is stored N-major (B(k,n) = Bc[n*ldb + k]):
// lane holds column n=lane&15, K = {kb..kb+7, 16+kb..16+kb+7} (2 x 16 B per lane).
__device__ __forceinline__ v16h load_b_frag_cm(const _Float16* __restrict__ Bc,
                                               int ldb, int lane) {
  const int n  = lane & 15;
  const int kb = (lane & 16) ? 8 : 0;
  const _Float16* col = Bc + (size_t)n * ldb + kb;
  v16h b;
#pragma unroll
  for (int i = 0; i < 16; ++i) b[i] = col[(i & 7) + ((i >> 3) << 4)];
  return b;
}

__device__ __forceinline__ v8f wmma_f16(v16h a, v16h b, v8f c) {
  return __builtin_amdgcn_wmma_f32_16x16x32_f16(false, a, false, b,
                                                (short)0, c, false, false);
}

__device__ __forceinline__ v8f zero8() {
  v8f z = {0.f, 0.f, 0.f, 0.f, 0.f, 0.f, 0.f, 0.f};
  return z;
}

// ---------------------------------------------------------------------------
// Kernel 1: embedding gather + LayerNorm.  One block per (b, l) row.
// ---------------------------------------------------------------------------
__global__ __launch_bounds__(128)
void embed_ln(const int* __restrict__ tok, const float* __restrict__ emb,
              const float* __restrict__ g, const float* __restrict__ bta,
              float* __restrict__ embOut, _Float16* __restrict__ normOut,
              float* __restrict__ maskb, int L, int Lp) {
  __shared__ float rs[128], rq[128];
  __shared__ float s_mu, s_rstd;
  const int blk = blockIdx.x;           // b*Lp + l
  const int b = blk / Lp, l = blk % Lp;
  const int t = threadIdx.x;
  const size_t rowOut = (size_t)blk * DM;

  if (l >= L) {
    for (int i = t; i < DM; i += 128) {
      embOut[rowOut + i]  = 0.f;
      normOut[rowOut + i] = (_Float16)0.f;
    }
    if (t == 0) maskb[blk] = -1.0e18f;
    return;
  }
  const int token = tok[(size_t)l * BATCH + b];   // tokens are [L, BATCH]
  if (t == 0) maskb[blk] = (token == PAD_IDX) ? -1.0e18f : 0.0f;

  const float* er = emb + (size_t)token * DM;
  float x[4], sum = 0.f, sq = 0.f;
#pragma unroll
  for (int i = 0; i < 4; ++i) {
    x[i] = er[t * 4 + i];
    sum += x[i];
    sq  += x[i] * x[i];
  }
  rs[t] = sum; rq[t] = sq;
  __syncthreads();
  for (int s = 64; s > 0; s >>= 1) {
    if (t < s) { rs[t] += rs[t + s]; rq[t] += rq[t + s]; }
    __syncthreads();
  }
  if (t == 0) {
    const float mu = rs[0] / DM;
    const float var = rq[0] / DM - mu * mu;
    s_mu = mu;
    s_rstd = rsqrtf(var + 1e-6f);
  }
  __syncthreads();
  const float mu = s_mu, rstd = s_rstd;
#pragma unroll
  for (int i = 0; i < 4; ++i) {
    const int d = t * 4 + i;
    embOut[rowOut + d]  = x[i];
    normOut[rowOut + d] = (_Float16)((x[i] - mu) * rstd * g[d] + bta[d]);
  }
}

// ---------------------------------------------------------------------------
// Kernel 2: weight conversion f32 -> f16 WITH transpose:  Wt[n,k] = W[k,n].
// Makes every GEMM B-operand K-contiguous per lane.
// ---------------------------------------------------------------------------
__global__ void w_to_f16_t(const float* __restrict__ in,
                           _Float16* __restrict__ out) {
  const int i = blockIdx.x * 256 + threadIdx.x;   // n*512 + k
  if (i >= DM * DM) return;
  const int n = i >> 9, k = i & (DM - 1);
  out[i] = (_Float16)in[(size_t)k * DM + n];
}

// ---------------------------------------------------------------------------
// Kernel 3: LDS-staged WMMA GEMM, C[M,512] = A[M,512] @ W[512,512]
//   block = 256 threads = 8 waves; block tile 128(M) x 64(N); K-step 32.
//   Weights are pre-transposed f16: Wt[n*512+k].
//   Staging uses the CDNA5 ASYNC pipe (global_load_async_to_lds_b128).
//   MODE 0: out = (acc + bias) * scale  -> f16 head layout     [B,H,Lp,64]
//   MODE 2: out =  acc + bias           -> f16 head-T layout   [B,H,64,Lp]
//   MODE 1: out =  acc + bias + resid   -> f32 [M,512], f16 [M,512],
//                                          f16 transposed [B,512,Lp]
// ---------------------------------------------------------------------------
template <int MODE>
__global__ __launch_bounds__(256)
void gemm512_wmma(const _Float16* __restrict__ A, const _Float16* __restrict__ Wt,
                  const float* __restrict__ bias, const float* __restrict__ resid,
                  _Float16* __restrict__ outF16, float* __restrict__ outF32,
                  _Float16* __restrict__ outF16T, int Lp, float scale) {
  __shared__ _Float16 As[128][40];   // 32 K-halfs + pad; row stride 80 B (16B-aligned)
  __shared__ _Float16 Bs[64][40];    // N-major: Bs[n][k]; row stride 80 B
  const int tid  = threadIdx.x;
  const int lane = tid & 31;
  const int w    = tid >> 5;
  const int n0   = blockIdx.x * 64;
  const int m0   = blockIdx.y * 128;

  v8f acc[4];
#pragma unroll
  for (int j = 0; j < 4; ++j) acc[j] = zero8();

  for (int kt = 0; kt < 16; ++kt) {
    const int k0 = kt * 32;
    // stage A tile 128x32 halfs = 512 x 16B chunks (2 per thread), async DMA
#pragma unroll
    for (int i = 0; i < 2; ++i) {
      const int idx = tid * 2 + i;           // chunk id
      const int r = idx >> 2, c = idx & 3;   // 4 chunks per 64 B row
      async_g2l_b128(&A[(size_t)(m0 + r) * DM + k0 + c * 8], &As[r][c * 8]);
    }
    // stage Wt tile 64(n) x 32(k) halfs = 256 x 16B chunks (1 per thread)
    {
      const int r = tid >> 2, c = tid & 3;
      async_g2l_b128(&Wt[(size_t)(n0 + r) * DM + k0 + c * 8], &Bs[r][c * 8]);
    }
    if (kt < 15)   // gfx1250 global_prefetch of the next K tile
      __builtin_prefetch(&A[(size_t)(m0 + (tid >> 1)) * DM + k0 + 32], 0, 1);
    wait_async_all();          // s_wait_asynccnt 0 (own wave's DMAs done)
    __syncthreads();           // all waves' DMAs done -> LDS tile valid

    const v16h a = load_a_frag(&As[w * 16][0], 40, lane);
#pragma unroll
    for (int j = 0; j < 4; ++j) {
      const v16h b = load_b_frag_cm(&Bs[j * 16][0], 40, lane);
      acc[j] = wmma_f16(a, b, acc[j]);
    }
    __syncthreads();
  }

  const int mw = m0 + w * 16;
  const int nn = lane & 15;
  const int hi = (lane >> 4) << 3;
#pragma unroll
  for (int j = 0; j < 4; ++j) {
    const int gn = n0 + j * 16 + nn;
#pragma unroll
    for (int r = 0; r < 8; ++r) {
      const int gm = mw + r + hi;
      float v = acc[j][r];
      const int b = gm / Lp, l = gm % Lp;
      if (MODE == 0) {
        v = (v + bias[gn]) * scale;
        const int h = gn >> 6, d = gn & 63;
        outF16[((size_t)(b * HEADS + h) * Lp + l) * HD + d] = (_Float16)v;
      } else if (MODE == 2) {
        v = v + bias[gn];
        const int h = gn >> 6, d = gn & 63;
        outF16[((size_t)(b * HEADS + h) * HD + d) * Lp + l] = (_Float16)v;
      } else {
        v = v + bias[gn] + resid[(size_t)gm * DM + gn];
        outF32[(size_t)gm * DM + gn] = v;
        outF16[(size_t)gm * DM + gn] = (_Float16)v;
        outF16T[((size_t)b * DM + gn) * Lp + l] = (_Float16)v;
      }
    }
  }
}

// ---------------------------------------------------------------------------
// Kernel 4: per-head scores = Q @ K^T + maskbias.   One wave per 16x32 tile.
//   Q, K: [B*H, Lp, 64] f16 (Q pre-scaled by 1/8).  scores: [B*H, Lp, Lp] f32
// ---------------------------------------------------------------------------
__global__ __launch_bounds__(32)
void attn_scores_wmma(const _Float16* __restrict__ Q, const _Float16* __restrict__ K,
                      const float* __restrict__ maskb, float* __restrict__ scores,
                      int Lp) {
  const int lane = threadIdx.x & 31;
  const int n0 = blockIdx.x * 32;
  const int m0 = blockIdx.y * 16;
  const size_t z = blockIdx.z;            // b*H + h
  const int bb = (int)(z >> 3);
  const _Float16* Qb = Q + ((size_t)z * Lp + m0) * HD;
  v8f acc0 = zero8(), acc1 = zero8();
#pragma unroll
  for (int kt = 0; kt < 2; ++kt) {
    const int k0 = kt * 32;
    const v16h a  = load_a_frag(Qb + k0, HD, lane);
    const v16h b0 = load_b_frag_cm(K + ((size_t)z * Lp + n0) * HD + k0, HD, lane);
    const v16h b1 = load_b_frag_cm(K + ((size_t)z * Lp + n0 + 16) * HD + k0, HD, lane);
    acc0 = wmma_f16(a, b0, acc0);
    acc1 = wmma_f16(a, b1, acc1);
  }
  const int nn = lane & 15;
  const int hi = (lane >> 4) << 3;
#pragma unroll
  for (int r = 0; r < 8; ++r) {
    const int gm = m0 + r + hi;
    const int g0 = n0 + nn, g1 = n0 + 16 + nn;
    scores[((size_t)z * Lp + gm) * Lp + g0] = acc0[r] + maskb[bb * Lp + g0];
    scores[((size_t)z * Lp + gm) * Lp + g1] = acc1[r] + maskb[bb * Lp + g1];
  }
}

// ---------------------------------------------------------------------------
// Kernel 5: row softmax, f32 scores -> f16 attn.  One block per row, Lp<=224.
// ---------------------------------------------------------------------------
__global__ __launch_bounds__(256)
void softmax_rows(const float* __restrict__ scores, _Float16* __restrict__ attn,
                  int Lp) {
  __shared__ float red[256];
  const size_t row = blockIdx.x;
  const int t = threadIdx.x;
  const float x = (t < Lp) ? scores[row * Lp + t] : -3.0e38f;
  red[t] = x;
  __syncthreads();
  for (int s = 128; s > 0; s >>= 1) {
    if (t < s) red[t] = fmaxf(red[t], red[t + s]);
    __syncthreads();
  }
  const float mx = red[0];
  __syncthreads();
  const float e = (t < Lp) ? __expf(x - mx) : 0.0f;
  red[t] = e;
  __syncthreads();
  for (int s = 128; s > 0; s >>= 1) {
    if (t < s) red[t] += red[t + s];
    __syncthreads();
  }
  const float inv = 1.0f / red[0];
  if (t < Lp) attn[row * Lp + t] = (_Float16)(e * inv);
}

// ---------------------------------------------------------------------------
// Kernel 6: ctx = attn @ V, relayout to [B, Lp, 512] f16.
//   V is head-transposed: Vt[B*H, 64, Lp] -> B frag is K-contiguous.
//   One wave per 16(M) x 64(N=head-dim) tile.
// ---------------------------------------------------------------------------
__global__ __launch_bounds__(32)
void attn_ctx_wmma(const _Float16* __restrict__ attn, const _Float16* __restrict__ Vt,
                   _Float16* __restrict__ ctx, int Lp) {
  const int lane = threadIdx.x & 31;
  const int m0 = blockIdx.y * 16;
  const size_t z = blockIdx.z;
  const int bb = (int)(z >> 3), h = (int)(z & 7);
  v8f acc[4];
#pragma unroll
  for (int j = 0; j < 4; ++j) acc[j] = zero8();
  const _Float16* Ab = attn + ((size_t)z * Lp + m0) * Lp;
  const _Float16* Bb = Vt + (size_t)z * HD * Lp;
  const int ks = Lp >> 5;
  for (int kt = 0; kt < ks; ++kt) {
    const int k0 = kt * 32;
    const v16h a = load_a_frag(Ab + k0, Lp, lane);
#pragma unroll
    for (int j = 0; j < 4; ++j) {
      const v16h b = load_b_frag_cm(Bb + (size_t)(j * 16) * Lp + k0, Lp, lane);
      acc[j] = wmma_f16(a, b, acc[j]);
    }
  }
  const int nn = lane & 15;
  const int hi = (lane >> 4) << 3;
#pragma unroll
  for (int j = 0; j < 4; ++j)
#pragma unroll
    for (int r = 0; r < 8; ++r) {
      const int gm = m0 + r + hi;
      const int gn = j * 16 + nn;
      ctx[((size_t)bb * Lp + gm) * DM + h * HD + gn] = (_Float16)acc[j][r];
    }
}

// ---------------------------------------------------------------------------
// Kernel 7: copy his_word [B,160,512] f32 -> d_out [B,150,512]
// ---------------------------------------------------------------------------
__global__ void copy_hisword(const float* __restrict__ wordh, float* __restrict__ out) {
  const size_t i = (size_t)blockIdx.x * 256 + threadIdx.x;
  if (i >= (size_t)BATCH * LH * DM) return;
  const int d = (int)(i & (DM - 1));
  const size_t r = i >> 9;
  const int l = (int)(r % LH), b = (int)(r / LH);
  out[i] = wordh[((size_t)b * LHP + l) * DM + d];
}

// ---------------------------------------------------------------------------
// Kernel 8: avg-pool his_word over 50-token utterances -> his_utt [B,3,512]
// ---------------------------------------------------------------------------
__global__ void avgpool_utt(const float* __restrict__ wordh, float* __restrict__ utt) {
  const int i = blockIdx.x * 256 + threadIdx.x;
  if (i >= BATCH * 3 * DM) return;
  const int d = i & (DM - 1);
  const int u = (i >> 9) % 3;
  const int b = i / (3 * DM);
  float s = 0.f;
  for (int k = 0; k < UTT; ++k)
    s += wordh[((size_t)b * LHP + u * UTT + k) * DM + d];
  utt[i] = s * (1.0f / UTT);
}

// ---------------------------------------------------------------------------
// Kernel 9: score_word = doc_word @ his_word^T.  One wave per 16x32 tile.
//   docw f16 [B,224,512]; hisw f16 [B,160,512] (N-major for B); sw f32 [B,224,160]
// ---------------------------------------------------------------------------
__global__ __launch_bounds__(32)
void score_word_wmma(const _Float16* __restrict__ docw,
                     const _Float16* __restrict__ hisw, float* __restrict__ sw) {
  const int lane = threadIdx.x & 31;
  const int n0 = blockIdx.x * 32;
  const int m0 = blockIdx.y * 16;
  const int b = blockIdx.z;
  v8f acc0 = zero8(), acc1 = zero8();
  const _Float16* Ab = docw + ((size_t)b * LDP + m0) * DM;
  const _Float16* Bb = hisw + (size_t)b * LHP * DM;
  for (int kt = 0; kt < 16; ++kt) {
    const int k0 = kt * 32;
    const v16h a  = load_a_frag(Ab + k0, DM, lane);
    const v16h b0 = load_b_frag_cm(Bb + (size_t)n0 * DM + k0, DM, lane);
    const v16h b1 = load_b_frag_cm(Bb + (size_t)(n0 + 16) * DM + k0, DM, lane);
    acc0 = wmma_f16(a, b0, acc0);
    acc1 = wmma_f16(a, b1, acc1);
  }
  const int nn = lane & 15;
  const int hi = (lane >> 4) << 3;
#pragma unroll
  for (int r = 0; r < 8; ++r) {
    const int gm = m0 + r + hi;
    sw[((size_t)b * LDP + gm) * LHP + n0 + nn]      = acc0[r];
    sw[((size_t)b * LDP + gm) * LHP + n0 + 16 + nn] = acc1[r];
  }
}

// ---------------------------------------------------------------------------
// Kernel 10: score_utt (3 dots of 512) fused with combine:
//   score[b,q,k] = score_word[b,q,k] * score_utt[b,q,k/50]   (k<150, q<200)
//   written zero-padded f16 [B,224,160] for the final WMMA GEMM.
// ---------------------------------------------------------------------------
__global__ __launch_bounds__(256)
void combine_scores(const float* __restrict__ docw, const float* __restrict__ utt,
                    const float* __restrict__ sw, _Float16* __restrict__ sc) {
  __shared__ float red[256];
  __shared__ float su[3];
  const int blk = blockIdx.x;          // b*LDP + q
  const int b = blk / LDP, q = blk % LDP;
  const int t = threadIdx.x;
  if (q < LD) {
    const float* dr = docw + (size_t)blk * DM;
    for (int u = 0; u < 3; ++u) {
      const float* ur = utt + ((size_t)b * 3 + u) * DM;
      red[t] = dr[t] * ur[t] + dr[t + 256] * ur[t + 256];
      __syncthreads();
      for (int s = 128; s > 0; s >>= 1) {
        if (t < s) red[t] += red[t + s];
        __syncthreads();
      }
      if (t == 0) su[u] = red[0];
      __syncthreads();
    }
  }
  if (t < LHP) {
    float v = 0.f;
    if (q < LD && t < LH) v = sw[(size_t)blk * LHP + t] * su[t / UTT];
    sc[(size_t)blk * LHP + t] = (_Float16)v;
  }
}

// ---------------------------------------------------------------------------
// Kernel 11: n_doc = score @ his_word -> d_out (f32).  Wave per 16x64 tile.
//   sc f16 [B,224,160]; hiswT f16 [B,512,160] (N-major); out f32 [B,200,512]
// ---------------------------------------------------------------------------
__global__ __launch_bounds__(32)
void ndoc_wmma(const _Float16* __restrict__ sc, const _Float16* __restrict__ hiswT,
               float* __restrict__ out) {
  const int lane = threadIdx.x & 31;
  const int n0 = blockIdx.x * 64;
  const int m0 = blockIdx.y * 16;
  const int b = blockIdx.z;
  v8f acc[4];
#pragma unroll
  for (int j = 0; j < 4; ++j) acc[j] = zero8();
  const _Float16* Ab = sc + ((size_t)b * LDP + m0) * LHP;
  const _Float16* Bb = hiswT + (size_t)b * DM * LHP;
#pragma unroll
  for (int kt = 0; kt < 5; ++kt) {         // K = 160 = 5 * 32
    const int k0 = kt * 32;
    const v16h a = load_a_frag(Ab + k0, LHP, lane);
#pragma unroll
    for (int j = 0; j < 4; ++j) {
      const v16h bf =
          load_b_frag_cm(Bb + (size_t)(n0 + j * 16) * LHP + k0, LHP, lane);
      acc[j] = wmma_f16(a, bf, acc[j]);
    }
  }
  const int nn = lane & 15;
  const int hi = (lane >> 4) << 3;
#pragma unroll
  for (int j = 0; j < 4; ++j)
#pragma unroll
    for (int r = 0; r < 8; ++r) {
      const int gm = m0 + r + hi;
      if (gm < LD)
        out[((size_t)b * LD + gm) * DM + n0 + j * 16 + nn] = acc[j][r];
    }
}

// ---------------------------------------------------------------------------
// Host-side orchestration
// ---------------------------------------------------------------------------
extern "C" void kernel_launch(void* const* d_in, const int* in_sizes, int n_in,
                              void* d_out, int out_size, void* d_ws, size_t ws_size,
                              hipStream_t stream) {
  (void)in_sizes; (void)n_in; (void)out_size; (void)ws_size;
  const int*   his  = (const int*)d_in[0];
  const int*   doc  = (const int*)d_in[1];
  const float* emb  = (const float*)d_in[2];
  const float* ln1g = (const float*)d_in[3];
  const float* ln1b = (const float*)d_in[4];
  const float* ln2g = (const float*)d_in[5];
  const float* ln2b = (const float*)d_in[6];
  const float* Wq = (const float*)d_in[7];  const float* bq = (const float*)d_in[8];
  const float* Wk = (const float*)d_in[9];  const float* bk = (const float*)d_in[10];
  const float* Wv = (const float*)d_in[11]; const float* bv = (const float*)d_in[12];
  const float* Wo = (const float*)d_in[13]; const float* bo = (const float*)d_in[14];

  char* ws = (char*)d_ws;
  size_t off = 0;
  auto carve = [&](size_t bytes) -> void* {
    void* p = ws + off;
    off = (off + bytes + 255) & ~(size_t)255;
    return p;
  };

  const size_t rowsD = (size_t)BATCH * LDP;            // max rows (doc phase)
  // Buffers shared between his/doc phases (phases run sequentially on stream):
  _Float16* normBuf = (_Float16*)carve(rowsD * DM * sizeof(_Float16));
  float*    embRes  = (float*)   carve(rowsD * DM * sizeof(float));
  float*    maskb   = (float*)   carve(rowsD * sizeof(float));
  _Float16* WqT = (_Float16*)carve((size_t)DM * DM * 2);   // transposed f16
  _Float16* WkT = (_Float16*)carve((size_t)DM * DM * 2);
  _Float16* WvT = (_Float16*)carve((size_t)DM * DM * 2);
  _Float16* WoT = (_Float16*)carve((size_t)DM * DM * 2);
  _Float16* Qh = (_Float16*)carve(rowsD * DM * 2);     // [B,H,Lp,64]
  _Float16* Kh = (_Float16*)carve(rowsD * DM * 2);     // [B,H,Lp,64]
  _Float16* Vt = (_Float16*)carve(rowsD * DM * 2);     // [B,H,64,Lp]
  float*    scores = (float*)carve((size_t)BATCH * HEADS * LDP * LDP * sizeof(float));
  _Float16* attnB  = (_Float16*)carve((size_t)BATCH * HEADS * LDP * LDP * 2);
  _Float16* ctxB   = (_Float16*)carve(rowsD * DM * 2);
  // Persistent outputs of the two phases:
  float*    wordHf32  = (float*)   carve((size_t)BATCH * LHP * DM * sizeof(float));
  _Float16* wordHf16  = (_Float16*)carve((size_t)BATCH * LHP * DM * 2);
  _Float16* wordHf16T = (_Float16*)carve((size_t)BATCH * LHP * DM * 2);
  float*    wordDf32  = (float*)   carve((size_t)BATCH * LDP * DM * sizeof(float));
  _Float16* wordDf16  = (_Float16*)carve((size_t)BATCH * LDP * DM * 2);
  _Float16* wordDf16T = (_Float16*)carve((size_t)BATCH * LDP * DM * 2);
  float*    utt = (float*)carve((size_t)BATCH * 3 * DM * sizeof(float));
  float*    sw  = (float*)carve((size_t)BATCH * LDP * LHP * sizeof(float));
  _Float16* scF = (_Float16*)carve((size_t)BATCH * LDP * LHP * 2);

  // Convert + transpose weights to f16 once per launch
  const int wn = DM * DM;
  w_to_f16_t<<<(wn + 255) / 256, 256, 0, stream>>>(Wq, WqT);
  w_to_f16_t<<<(wn + 255) / 256, 256, 0, stream>>>(Wk, WkT);
  w_to_f16_t<<<(wn + 255) / 256, 256, 0, stream>>>(Wv, WvT);
  w_to_f16_t<<<(wn + 255) / 256, 256, 0, stream>>>(Wo, WoT);

  auto run_phase = [&](const int* tok, const float* lng, const float* lnb,
                       int L, int Lp, float* wordF32, _Float16* wordF16,
                       _Float16* wordF16T) {
    const int rows = BATCH * Lp;
    embed_ln<<<rows, 128, 0, stream>>>(tok, emb, lng, lnb, embRes, normBuf,
                                       maskb, L, Lp);
    gemm512_wmma<0><<<dim3(8, rows / 128), 256, 0, stream>>>(
        normBuf, WqT, bq, nullptr, Qh, nullptr, nullptr, Lp, 0.125f); // 1/sqrt(64)
    gemm512_wmma<0><<<dim3(8, rows / 128), 256, 0, stream>>>(
        normBuf, WkT, bk, nullptr, Kh, nullptr, nullptr, Lp, 1.0f);
    gemm512_wmma<2><<<dim3(8, rows / 128), 256, 0, stream>>>(
        normBuf, WvT, bv, nullptr, Vt, nullptr, nullptr, Lp, 1.0f);
    attn_scores_wmma<<<dim3(Lp / 32, Lp / 16, BATCH * HEADS), 32, 0, stream>>>(
        Qh, Kh, maskb, scores, Lp);
    softmax_rows<<<BATCH * HEADS * Lp, 256, 0, stream>>>(scores, attnB, Lp);
    attn_ctx_wmma<<<dim3(1, Lp / 16, BATCH * HEADS), 32, 0, stream>>>(
        attnB, Vt, ctxB, Lp);
    gemm512_wmma<1><<<dim3(8, rows / 128), 256, 0, stream>>>(
        ctxB, WoT, bo, embRes, wordF16, wordF32, wordF16T, Lp, 1.0f);
  };

  // --- history phase ---
  run_phase(his, ln1g, ln1b, LH, LHP, wordHf32, wordHf16, wordHf16T);
  {
    const size_t n = (size_t)BATCH * LH * DM;
    copy_hisword<<<(unsigned)((n + 255) / 256), 256, 0, stream>>>(
        wordHf32, (float*)d_out);
    avgpool_utt<<<(BATCH * 3 * DM + 255) / 256, 256, 0, stream>>>(wordHf32, utt);
  }

  // --- document phase ---
  run_phase(doc, ln2g, ln2b, LD, LDP, wordDf32, wordDf16, wordDf16T);

  // --- cross scoring ---
  score_word_wmma<<<dim3(LHP / 32, LDP / 16, BATCH), 32, 0, stream>>>(
      wordDf16, wordHf16, sw);
  combine_scores<<<BATCH * LDP, 256, 0, stream>>>(wordDf32, utt, sw, scF);
  ndoc_wmma<<<dim3(DM / 64, LDP / 16, BATCH), 32, 0, stream>>>(
      scF, wordHf16T, (float*)d_out + (size_t)BATCH * LH * DM);
}
